// GATGCN_58729382805579
// MI455X (gfx1250) — matrix operand order repaired and checked by reading, not compiled, not run on hardware
//
#include <hip/hip_runtime.h>
#include <hip/hip_bf16.h>
#include <math.h>

// ---------------- problem constants (from reference) ----------------
#define N_NODES   20000
#define E_EDGES   240000
#define ET_EDGES  (E_EDGES + N_NODES)   // + self loops
#define F_IN      768
#define H1_DIM    128
#define PE_DIM    32
#define D_DIM     160                   // H1 + PE
#define HEADS     5
#define HC_DIM    800                   // HEADS * D
#define H2_DIM    128
#define OUT_DIM   16
#define G_GRAPHS  64
#define NEG_SLOPE 0.2f

typedef __bf16 bf16_t;
typedef __attribute__((ext_vector_type(16))) __bf16 v16bf;
typedef __attribute__((ext_vector_type(8)))  __bf16 v8bf;
typedef __attribute__((ext_vector_type(8)))  float  v8f;

// round-to-nearest-even f32 -> bf16 (bit-level, no __bf16 conversion dependence)
__device__ __forceinline__ bf16_t f2bf(float f) {
  unsigned u = __float_as_uint(f);
  u = (u + 0x7FFFu + ((u >> 16) & 1u)) >> 16;
  unsigned short s = (unsigned short)u;
  bf16_t b;
  __builtin_memcpy(&b, &s, sizeof(b));
  return b;
}

__device__ __forceinline__ void atomicMaxF(float* addr, float val) {
  unsigned int* ua = (unsigned int*)addr;
  unsigned int old = *ua;
  for (;;) {
    float cur = __uint_as_float(old);
    if (cur >= val) break;
    unsigned int assumed = old;
    old = atomicCAS(ua, assumed, __float_as_uint(val));
    if (old == assumed) break;
  }
}

// ---------------- utility kernels ----------------
__global__ void k_fill(float* __restrict__ p, float v, long long n) {
  long long i = (long long)blockIdx.x * blockDim.x + threadIdx.x;
  long long stride = (long long)gridDim.x * blockDim.x;
  for (; i < n; i += stride) p[i] = v;
}

__global__ void k_f32_to_bf16(const float* __restrict__ in, bf16_t* __restrict__ out,
                              long long n) {
  long long i = (long long)blockIdx.x * blockDim.x + threadIdx.x;
  long long stride = (long long)gridDim.x * blockDim.x;
  for (; i < n; i += stride) out[i] = f2bf(in[i]);
}

// Wt[c*rows + r] = (bf16) W[r*cols + c]   (W is [rows, cols] row-major)
__global__ void k_transpose_bf16(const float* __restrict__ W, bf16_t* __restrict__ Wt,
                                 int rows, int cols) {
  long long i = (long long)blockIdx.x * blockDim.x + threadIdx.x;
  long long n = (long long)rows * cols;
  if (i >= n) return;
  int c = (int)(i / rows);
  int r = (int)(i % rows);
  Wt[(size_t)c * rows + r] = f2bf(W[(size_t)r * cols + c]);
}

// h[:, 128:160] = pe_enc (bf16)
__global__ void k_copy_pe(const float* __restrict__ pe, bf16_t* __restrict__ h) {
  long long i = (long long)blockIdx.x * blockDim.x + threadIdx.x;
  if (i >= (long long)N_NODES * PE_DIM) return;
  int n = (int)(i / PE_DIM);
  int j = (int)(i % PE_DIM);
  h[(size_t)n * D_DIM + H1_DIM + j] = f2bf(pe[i]);
}

// ---------------- WMMA bf16 GEMM, 32x32 macro-tile per wave ----------------
// C[M, Ncols] (ld = ldc) = act( A[M,K](bf16, row-major) * Bt[Ncols,K]^T + bias )
// Each wave owns a 32x32 output block: 2 A fragments x 2 B fragments ->
// 4 f32 accumulators, 4x v_wmma_f32_16x16x32_bf16 per 32-wide K step.
// Doubles L2 arithmetic intensity vs one-tile-per-wave (~32 FLOP/B).
// Requires: M % 32 == 0, Ncols % 32 == 0, K % 32 == 0 (true for all uses here).
__global__ void k_wmma_gemm(const bf16_t* __restrict__ A,
                            const bf16_t* __restrict__ Bt,
                            const float* __restrict__ bias,
                            float* __restrict__ Cf, bf16_t* __restrict__ Cb,
                            int M, int K, int Ncols, int ldc, int do_relu) {
  int wave = (blockIdx.x * blockDim.x + threadIdx.x) >> 5;  // wave-uniform
  int lane = threadIdx.x & 31;
  int nct = Ncols >> 5;
  int total = (M >> 5) * nct;
  if (wave >= total) return;  // whole wave exits -> EXEC all-1s for WMMA
  int rt = wave / nct;
  int ct = wave - rt * nct;
  int row0 = rt << 5;
  int col0 = ct << 5;

  int lo = lane & 15;
  int hi = lane >> 4;

  // A fragment: lanes 0-15 hold row=lo, K = {0..15}; lanes 16-31 same
  // row with K = {16..31}  (contiguous, matches B)
  const bf16_t* arow0 = A + (size_t)(row0 + lo) * K + (hi << 4);
  const bf16_t* arow1 = arow0 + (size_t)16 * K;
  // B fragment: lane holds column col0+lo, contiguous 16 K-values offset by
  // 16*hi (same k order as A) -> one 32-byte load from transposed weights
  const bf16_t* brow0 = Bt + (size_t)(col0 + lo) * K + (hi << 4);
  const bf16_t* brow1 = brow0 + (size_t)16 * K;

  v8f acc00, acc01, acc10, acc11;
  for (int i = 0; i < 8; ++i) {
    acc00[i] = 0.0f; acc01[i] = 0.0f; acc10[i] = 0.0f; acc11[i] = 0.0f;
  }

  for (int k0 = 0; k0 < K; k0 += 32) {
    // prefetch activation stream 2 K-steps ahead (global_prefetch_b8)
    if (k0 + 64 < K) {
      __builtin_prefetch(arow0 + k0 + 64, 0, 1);
      __builtin_prefetch(arow1 + k0 + 64, 0, 1);
    }
    v8bf a00 = *(const v8bf*)(arow0 + k0);
    v8bf a01 = *(const v8bf*)(arow0 + k0 + 8);
    v8bf a10 = *(const v8bf*)(arow1 + k0);
    v8bf a11 = *(const v8bf*)(arow1 + k0 + 8);
    v16bf b0 = *(const v16bf*)(brow0 + k0);
    v16bf b1 = *(const v16bf*)(brow1 + k0);
    v16bf a0, a1;
    for (int i = 0; i < 8; ++i) {
      a0[i] = a00[i]; a0[i + 8] = a01[i];
      a1[i] = a10[i]; a1[i + 8] = a11[i];
    }
    acc00 = __builtin_amdgcn_wmma_f32_16x16x32_bf16(false, a0, false, b0,
                                                    (short)0, acc00, false, false);
    acc01 = __builtin_amdgcn_wmma_f32_16x16x32_bf16(false, a0, false, b1,
                                                    (short)0, acc01, false, false);
    acc10 = __builtin_amdgcn_wmma_f32_16x16x32_bf16(false, a1, false, b0,
                                                    (short)0, acc10, false, false);
    acc11 = __builtin_amdgcn_wmma_f32_16x16x32_bf16(false, a1, false, b1,
                                                    (short)0, acc11, false, false);
  }

  // D layout: elem v -> row = r0 + v + 8*hi, col = c0 + lo
  auto store_tile = [&](const v8f& acc, int r0, int c0) {
    int col = c0 + lo;
    float bv = bias ? bias[col] : 0.0f;
    for (int v = 0; v < 8; ++v) {
      int row = r0 + v + (hi << 3);
      float xv = acc[v] + bv;
      if (do_relu) xv = fmaxf(xv, 0.0f);
      if (Cf) Cf[(size_t)row * ldc + col] = xv;
      else    Cb[(size_t)row * ldc + col] = f2bf(xv);
    }
  };
  store_tile(acc00, row0,      col0);
  store_tile(acc01, row0,      col0 + 16);
  store_tile(acc10, row0 + 16, col0);
  store_tile(acc11, row0 + 16, col0 + 16);
}

// ---------------- GATv2 edge kernels ----------------
__device__ __forceinline__ void edge_ends(const int* __restrict__ ei, int e,
                                          int& src, int& dst) {
  if (e < E_EDGES) { src = ei[e]; dst = ei[E_EDGES + e]; }
  else             { src = e - E_EDGES; dst = src; }      // self loop
}

// score[e,h] = sum_d leaky(x_l[src] + x_r[dst]) * att[h]; track per-dst max
__global__ void k_edge_score(const float* __restrict__ xl, const float* __restrict__ xr,
                             const float* __restrict__ att, const int* __restrict__ ei,
                             float* __restrict__ score, float* __restrict__ smax) {
  long long i = (long long)blockIdx.x * blockDim.x + threadIdx.x;
  if (i >= (long long)ET_EDGES * HEADS) return;
  int e = (int)(i / HEADS);
  int h = (int)(i % HEADS);
  int src, dst; edge_ends(ei, e, src, dst);
  const float* pl = xl + (size_t)src * HC_DIM + h * D_DIM;
  const float* pr = xr + (size_t)dst * HC_DIM + h * D_DIM;
  const float* pa = att + h * D_DIM;
  float s = 0.0f;
  for (int d = 0; d < D_DIM; ++d) {
    float m = pl[d] + pr[d];
    m = (m > 0.0f) ? m : NEG_SLOPE * m;
    s += m * pa[d];
  }
  score[i] = s;
  atomicMaxF(&smax[(size_t)dst * HEADS + h], s);
}

// score <- exp(score - smax[dst]); den[dst] += exp
__global__ void k_edge_exp(float* __restrict__ score, const float* __restrict__ smax,
                           float* __restrict__ den, const int* __restrict__ ei) {
  long long i = (long long)blockIdx.x * blockDim.x + threadIdx.x;
  if (i >= (long long)ET_EDGES * HEADS) return;
  int e = (int)(i / HEADS);
  int h = (int)(i % HEADS);
  int src, dst; edge_ends(ei, e, src, dst);
  float ex = expf(score[i] - smax[(size_t)dst * HEADS + h]);
  score[i] = ex;
  atomicAdd(&den[(size_t)dst * HEADS + h], ex);
}

__global__ void k_edge_deg(const int* __restrict__ ei, float* __restrict__ deg) {
  int e = blockIdx.x * blockDim.x + threadIdx.x;
  if (e >= ET_EDGES) return;
  int src, dst; edge_ends(ei, e, src, dst);
  atomicAdd(&deg[dst], 1.0f);
}

// agg[dst,c] += x_l[src,c] * alpha[e, c/D]
__global__ void k_edge_agg(const float* __restrict__ xl, const float* __restrict__ ex,
                           const float* __restrict__ den, const int* __restrict__ ei,
                           float* __restrict__ agg) {
  long long i = (long long)blockIdx.x * blockDim.x + threadIdx.x;
  if (i >= (long long)ET_EDGES * HC_DIM) return;
  int e = (int)(i / HC_DIM);
  int c = (int)(i % HC_DIM);
  int h = c / D_DIM;
  int src, dst; edge_ends(ei, e, src, dst);
  float alpha = ex[(size_t)e * HEADS + h] / den[(size_t)dst * HEADS + h];
  atomicAdd(&agg[(size_t)dst * HC_DIM + c], xl[(size_t)src * HC_DIM + c] * alpha);
}

// g1 = relu(agg + gat_bias) as bf16 (feeds GCN GEMM)
__global__ void k_g1(const float* __restrict__ agg, const float* __restrict__ gat_bias,
                     bf16_t* __restrict__ g1) {
  long long i = (long long)blockIdx.x * blockDim.x + threadIdx.x;
  if (i >= (long long)N_NODES * HC_DIM) return;
  int c = (int)(i % HC_DIM);
  g1[i] = f2bf(fmaxf(agg[i] + gat_bias[c], 0.0f));
}

__global__ void k_dinv(const float* __restrict__ deg, float* __restrict__ dinv) {
  int n = blockIdx.x * blockDim.x + threadIdx.x;
  if (n >= N_NODES) return;
  float d = deg[n];
  dinv[n] = (d > 0.0f) ? rsqrtf(d) : 0.0f;
}

// g2[dst,c] += hg[src,c] * dinv[src]*dinv[dst]
__global__ void k_gcn_agg(const float* __restrict__ hg, const float* __restrict__ dinv,
                          const int* __restrict__ ei, float* __restrict__ g2) {
  long long i = (long long)blockIdx.x * blockDim.x + threadIdx.x;
  if (i >= (long long)ET_EDGES * HC_DIM) return;
  int e = (int)(i / HC_DIM);
  int c = (int)(i % HC_DIM);
  int src, dst; edge_ends(ei, e, src, dst);
  float norm = dinv[src] * dinv[dst];
  atomicAdd(&g2[(size_t)dst * HC_DIM + c], hg[(size_t)src * HC_DIM + c] * norm);
}

// v = relu(g2 + b_gcn); weighted-sum + max pooling per graph
__global__ void k_pool(const float* __restrict__ g2, const float* __restrict__ b_gcn,
                       const float* __restrict__ fw, const int* __restrict__ batch,
                       float* __restrict__ ps, float* __restrict__ wsum,
                       float* __restrict__ mx) {
  long long i = (long long)blockIdx.x * blockDim.x + threadIdx.x;
  if (i >= (long long)N_NODES * HC_DIM) return;
  int n = (int)(i / HC_DIM);
  int c = (int)(i % HC_DIM);
  float v = fmaxf(g2[i] + b_gcn[c], 0.0f);
  int g = batch[n];
  float w = fw[n];
  atomicAdd(&ps[(size_t)g * HC_DIM + c], v * w);
  atomicMaxF(&mx[(size_t)g * HC_DIM + c], v);  // v >= 0; mx init 0 matches ref
  if (c == 0) atomicAdd(&wsum[g], w);
}

// gf1 = relu([mean, max] @ W_fc + b_fc)
__global__ void k_fc(const float* __restrict__ ps, const float* __restrict__ wsum,
                     const float* __restrict__ mx, const float* __restrict__ W_fc,
                     const float* __restrict__ b_fc, float* __restrict__ gf1) {
  int i = blockIdx.x * blockDim.x + threadIdx.x;
  if (i >= G_GRAPHS * H2_DIM) return;
  int g = i / H2_DIM;
  int j = i % H2_DIM;
  float wsv = fmaxf(wsum[g], 1e-6f);
  float acc = b_fc[j];
  for (int c = 0; c < HC_DIM; ++c) {
    acc += (ps[(size_t)g * HC_DIM + c] / wsv) * W_fc[(size_t)c * H2_DIM + j];
    acc += mx[(size_t)g * HC_DIM + c] * W_fc[(size_t)(HC_DIM + c) * H2_DIM + j];
  }
  gf1[i] = fmaxf(acc, 0.0f);
}

__global__ void k_out(const float* __restrict__ gf1, const float* __restrict__ W_out,
                      const float* __restrict__ b_out, float* __restrict__ out) {
  int i = blockIdx.x * blockDim.x + threadIdx.x;
  if (i >= G_GRAPHS * OUT_DIM) return;
  int g = i / OUT_DIM;
  int o = i % OUT_DIM;
  float acc = b_out[o];
  for (int j = 0; j < H2_DIM; ++j)
    acc += gf1[(size_t)g * H2_DIM + j] * W_out[(size_t)j * OUT_DIM + o];
  out[i] = fmaxf(acc, 0.0f);
}

// ---------------- host side ----------------
static inline size_t alignup256(size_t x) { return (x + 255) & ~(size_t)255; }

extern "C" void kernel_launch(void* const* d_in, const int* in_sizes, int n_in,
                              void* d_out, int out_size, void* d_ws, size_t ws_size,
                              hipStream_t stream) {
  (void)in_sizes; (void)n_in; (void)out_size; (void)ws_size;
  const float* x        = (const float*)d_in[0];
  const float* pe_enc   = (const float*)d_in[1];
  const int*   eidx     = (const int*)d_in[2];
  const int*   batch    = (const int*)d_in[3];
  const float* flat_w   = (const float*)d_in[4];
  const float* W_pre    = (const float*)d_in[6];
  const float* b_pre    = (const float*)d_in[7];
  const float* W_l      = (const float*)d_in[8];
  const float* b_l      = (const float*)d_in[9];
  const float* W_r      = (const float*)d_in[10];
  const float* b_r      = (const float*)d_in[11];
  const float* att      = (const float*)d_in[12];
  const float* gat_bias = (const float*)d_in[13];
  const float* W_gcn    = (const float*)d_in[14];
  const float* b_gcn    = (const float*)d_in[15];
  const float* W_fc     = (const float*)d_in[16];
  const float* b_fc     = (const float*)d_in[17];
  const float* W_out    = (const float*)d_in[18];
  const float* b_out    = (const float*)d_in[19];
  float* out = (float*)d_out;

  // --- carve workspace ---
  char* base = (char*)d_ws;
  size_t off = 0;
  auto carve = [&](size_t bytes) -> void* {
    void* p = base + off;
    off += alignup256(bytes);
    return p;
  };
  bf16_t* xbf   = (bf16_t*)carve((size_t)N_NODES * F_IN * 2);
  bf16_t* wpreT = (bf16_t*)carve((size_t)H1_DIM * F_IN * 2);
  bf16_t* hbf   = (bf16_t*)carve((size_t)N_NODES * D_DIM * 2);
  bf16_t* wlT   = (bf16_t*)carve((size_t)HC_DIM * D_DIM * 2);
  bf16_t* wrT   = (bf16_t*)carve((size_t)HC_DIM * D_DIM * 2);
  bf16_t* wgcnT = (bf16_t*)carve((size_t)HC_DIM * HC_DIM * 2);
  float*  xl    = (float*)carve((size_t)N_NODES * HC_DIM * 4);
  float*  xr    = (float*)carve((size_t)N_NODES * HC_DIM * 4);
  float*  agg   = (float*)carve((size_t)N_NODES * HC_DIM * 4);
  float*  score = (float*)carve((size_t)ET_EDGES * HEADS * 4);
  float*  smax  = (float*)carve((size_t)N_NODES * HEADS * 4);
  float*  den   = (float*)carve((size_t)N_NODES * HEADS * 4);
  float*  deg   = (float*)carve((size_t)N_NODES * 4);
  float*  dinv  = (float*)carve((size_t)N_NODES * 4);
  float*  ps    = (float*)carve((size_t)G_GRAPHS * HC_DIM * 4);
  float*  wsum  = (float*)carve((size_t)G_GRAPHS * 4);
  float*  mx    = (float*)carve((size_t)G_GRAPHS * HC_DIM * 4);
  float*  gf1   = (float*)carve((size_t)G_GRAPHS * H2_DIM * 4);
  // aliases (lifetimes do not overlap)
  bf16_t* g1bf = (bf16_t*)xl;   // used after xl is dead
  float*  hg   = xr;            // used after xr is dead
  float*  g2   = agg;           // used after agg is dead

  const int T = 256;
  auto blocks = [](long long n, int t) { return (unsigned)((n + t - 1) / t); };

  // --- convert / transpose to bf16 ---
  k_f32_to_bf16<<<blocks((long long)N_NODES * F_IN, T), T, 0, stream>>>(
      x, xbf, (long long)N_NODES * F_IN);
  k_transpose_bf16<<<blocks((long long)F_IN * H1_DIM, T), T, 0, stream>>>(
      W_pre, wpreT, F_IN, H1_DIM);
  k_transpose_bf16<<<blocks((long long)D_DIM * HC_DIM, T), T, 0, stream>>>(
      W_l, wlT, D_DIM, HC_DIM);
  k_transpose_bf16<<<blocks((long long)D_DIM * HC_DIM, T), T, 0, stream>>>(
      W_r, wrT, D_DIM, HC_DIM);
  k_transpose_bf16<<<blocks((long long)HC_DIM * HC_DIM, T), T, 0, stream>>>(
      W_gcn, wgcnT, HC_DIM, HC_DIM);

  auto gemm_blocks = [](int M, int Nc) {
    long long waves = (long long)(M / 32) * (Nc / 32);  // 32x32 macro tiles
    return (unsigned)((waves + 7) / 8);                 // 8 waves / 256-thread block
  };

  // --- h[:, :128] = relu(x @ W_pre + b_pre); h[:, 128:] = pe ---
  k_wmma_gemm<<<gemm_blocks(N_NODES, H1_DIM), T, 0, stream>>>(
      xbf, wpreT, b_pre, (float*)nullptr, hbf,
      N_NODES, F_IN, H1_DIM, D_DIM, /*relu=*/1);
  k_copy_pe<<<blocks((long long)N_NODES * PE_DIM, T), T, 0, stream>>>(pe_enc, hbf);

  // --- x_l / x_r = h @ W_l|W_r + b ---
  k_wmma_gemm<<<gemm_blocks(N_NODES, HC_DIM), T, 0, stream>>>(
      hbf, wlT, b_l, xl, (bf16_t*)nullptr, N_NODES, D_DIM, HC_DIM, HC_DIM, 0);
  k_wmma_gemm<<<gemm_blocks(N_NODES, HC_DIM), T, 0, stream>>>(
      hbf, wrT, b_r, xr, (bf16_t*)nullptr, N_NODES, D_DIM, HC_DIM, HC_DIM, 0);

  // --- GATv2 edge softmax + aggregation ---
  k_fill<<<blocks((long long)N_NODES * HEADS, T), T, 0, stream>>>(
      smax, -3.0e38f, (long long)N_NODES * HEADS);
  k_fill<<<blocks((long long)N_NODES * HEADS, T), T, 0, stream>>>(
      den, 0.0f, (long long)N_NODES * HEADS);
  k_fill<<<blocks(N_NODES, T), T, 0, stream>>>(deg, 0.0f, N_NODES);
  k_fill<<<blocks((long long)N_NODES * HC_DIM, T), T, 0, stream>>>(
      agg, 0.0f, (long long)N_NODES * HC_DIM);

  k_edge_score<<<blocks((long long)ET_EDGES * HEADS, T), T, 0, stream>>>(
      xl, xr, att, eidx, score, smax);
  k_edge_exp<<<blocks((long long)ET_EDGES * HEADS, T), T, 0, stream>>>(
      score, smax, den, eidx);
  k_edge_deg<<<blocks(ET_EDGES, T), T, 0, stream>>>(eidx, deg);
  k_edge_agg<<<blocks((long long)ET_EDGES * HC_DIM, T), T, 0, stream>>>(
      xl, score, den, eidx, agg);

  // --- g1 = relu(agg + gat_bias) (bf16, aliases xl) ---
  k_g1<<<blocks((long long)N_NODES * HC_DIM, T), T, 0, stream>>>(agg, gat_bias, g1bf);

  // --- hg = g1 @ W_gcn (f32, aliases xr) ---
  k_wmma_gemm<<<gemm_blocks(N_NODES, HC_DIM), T, 0, stream>>>(
      g1bf, wgcnT, (const float*)nullptr, hg, (bf16_t*)nullptr,
      N_NODES, HC_DIM, HC_DIM, HC_DIM, 0);

  // --- GCN normalized aggregation (g2 aliases agg; rezero first) ---
  k_dinv<<<blocks(N_NODES, T), T, 0, stream>>>(deg, dinv);
  k_fill<<<blocks((long long)N_NODES * HC_DIM, T), T, 0, stream>>>(
      g2, 0.0f, (long long)N_NODES * HC_DIM);
  k_gcn_agg<<<blocks((long long)ET_EDGES * HC_DIM, T), T, 0, stream>>>(
      hg, dinv, eidx, g2);

  // --- pooling ---
  k_fill<<<blocks((long long)G_GRAPHS * HC_DIM, T), T, 0, stream>>>(
      ps, 0.0f, (long long)G_GRAPHS * HC_DIM);
  k_fill<<<blocks(G_GRAPHS, T), T, 0, stream>>>(wsum, 0.0f, G_GRAPHS);
  k_fill<<<blocks((long long)G_GRAPHS * HC_DIM, T), T, 0, stream>>>(
      mx, 0.0f, (long long)G_GRAPHS * HC_DIM);
  k_pool<<<blocks((long long)N_NODES * HC_DIM, T), T, 0, stream>>>(
      g2, b_gcn, flat_w, batch, ps, wsum, mx);

  // --- final MLP ---
  k_fc<<<blocks(G_GRAPHS * H2_DIM, T), T, 0, stream>>>(ps, wsum, mx, W_fc, b_fc, gf1);
  k_out<<<blocks(G_GRAPHS * OUT_DIM, T), T, 0, stream>>>(gf1, W_out, b_out, out);
}
